// SIC_87368224735692
// MI455X (gfx1250) — matrix operand order repaired and checked by reading, not compiled, not run on hardware
//
#include <hip/hip_runtime.h>
#include <hip/hip_bf16.h>

// Problem constants (match reference setup_inputs()).
#define BN 512       // batch rows (x)
#define SN 16384     // support rows (sx)
#define DN 2048      // feature dim
#define CN 1000      // classes
#define EPSF 1e-12f

// Block tile: M=64 (b), N=256 (s), Kc=32. 256 threads = 8 waves (wave32),
// waves arranged 4(M) x 2(N); each wave computes 16x128 via 8 f32 accumulators.
#define MT 64
#define NT 256
#define KC 32
#define LDA 40   // padded LDS row stride (bf16 elems) to avoid bank conflicts

typedef unsigned short u16;
typedef __attribute__((ext_vector_type(4)))  u16    u16x4;
typedef __attribute__((ext_vector_type(8)))  u16    u16x8;
typedef __attribute__((ext_vector_type(16))) u16    u16x16;
typedef __attribute__((ext_vector_type(4)))  float  f32x4;
typedef __attribute__((ext_vector_type(4)))  __bf16 bf16x4;
typedef __attribute__((ext_vector_type(16))) __bf16 v16bf;
typedef __attribute__((ext_vector_type(8)))  float  v8f;

// Split f32x4 -> (hi, lo) bf16 quads with RNE; hi+lo reconstructs ~16 bits.
// convertvector lets the backend use packed v_cvt_pk_bf16_f32.
__device__ __forceinline__ void split4(float4 v, u16x4& h, u16x4& l) {
    f32x4 f = {v.x, v.y, v.z, v.w};
    bf16x4 hb = __builtin_convertvector(f, bf16x4);
    f32x4  hf = __builtin_convertvector(hb, f32x4);
    bf16x4 lb = __builtin_convertvector(f - hf, bf16x4);
    h = __builtin_bit_cast(u16x4, hb);
    l = __builtin_bit_cast(u16x4, lb);
}

// Assemble a 16-element bf16 WMMA fragment from two 8-element LDS chunks.
__device__ __forceinline__ v16bf frag16(const u16* p0, const u16* p1) {
    u16x8 a = *(const u16x8*)p0;
    u16x8 b = *(const u16x8*)p1;
    u16x16 c = __builtin_shufflevector(a, b, 0,1,2,3,4,5,6,7,8,9,10,11,12,13,14,15);
    return __builtin_bit_cast(v16bf, c);
}

// ---- Pass 1a (big-ws path): split planes + reciprocal norms in one sweep ----
__global__ void SIC_prep_kernel(const float* __restrict__ x,
                                const float* __restrict__ sx,
                                u16* __restrict__ xh,  u16* __restrict__ xl,
                                u16* __restrict__ sxh, u16* __restrict__ sxl,
                                float* __restrict__ rinv, float* __restrict__ xinv) {
    const int blk = blockIdx.x;
    const float* in;
    u16 *oh, *ol;
    float* dst;
    if (blk < SN) {
        in = sx + (size_t)blk * DN;
        oh = sxh + (size_t)blk * DN; ol = sxl + (size_t)blk * DN;
        dst = rinv + blk;
    } else {
        int b = blk - SN;
        in = x + (size_t)b * DN;
        oh = xh + (size_t)b * DN; ol = xl + (size_t)b * DN;
        dst = xinv + b;
    }
    const int t = threadIdx.x;
    float ss = 0.0f;
    for (int i = t; i < DN / 4; i += 256) {
        float4 v = ((const float4*)in)[i];
        ss += v.x * v.x + v.y * v.y + v.z * v.z + v.w * v.w;
        u16x4 h, l;
        split4(v, h, l);
        *(u16x4*)(oh + i * 4) = h;
        *(u16x4*)(ol + i * 4) = l;
    }
    __shared__ float red[256];
    red[t] = ss;
    __syncthreads();
    for (int o = 128; o > 0; o >>= 1) {
        if (t < o) red[t] += red[t + o];
        __syncthreads();
    }
    if (t == 0) *dst = 1.0f / (sqrtf(red[0]) + EPSF);
}

// ---- Pass 1b (small-ws path): reciprocal norms only -------------------------
__global__ void SIC_norms_kernel(const float* __restrict__ x,
                                 const float* __restrict__ sx,
                                 float* __restrict__ rinv,
                                 float* __restrict__ xinv) {
    const int blk = blockIdx.x;
    const float* row;
    float* dst;
    if (blk < SN) { row = sx + (size_t)blk * DN;        dst = rinv + blk; }
    else          { row = x  + (size_t)(blk - SN) * DN; dst = xinv + (blk - SN); }
    const int t = threadIdx.x;
    float ss = 0.0f;
    for (int i = t; i < DN / 4; i += 256) {
        float4 v = ((const float4*)row)[i];
        ss += v.x * v.x + v.y * v.y + v.z * v.z + v.w * v.w;
    }
    __shared__ float red[256];
    red[t] = ss;
    __syncthreads();
    for (int o = 128; o > 0; o >>= 1) {
        if (t < o) red[t] += red[t + o];
        __syncthreads();
    }
    if (t == 0) *dst = 1.0f / (sqrtf(red[0]) + EPSF);
}

// ---- Pass 2: zero the output (harness poisons d_out) ------------------------
__global__ void SIC_zero_kernel(float* __restrict__ out, int n) {
    int i = blockIdx.x * blockDim.x + threadIdx.x;
    if (i < n) out[i] = 0.0f;
}

// ---- Pass 3: split-bf16 WMMA GEMM + score + scatter -------------------------
// PRE=true : stage pre-split bf16 planes (pure copies, ~zero staging VALU).
// PRE=false: stage f32 and split on the fly (small-workspace fallback).
template<bool PRE>
__global__ void __launch_bounds__(256)
SIC_gemm_kernel(const float* __restrict__ x,      // [BN, DN] f32
                const float* __restrict__ sx,     // [SN, DN] f32
                const u16*   __restrict__ xh,  const u16* __restrict__ xl,
                const u16*   __restrict__ sxh, const u16* __restrict__ sxl,
                const float* __restrict__ rinv,   // [SN]
                const float* __restrict__ xinv,   // [BN]
                const int*   __restrict__ sy,     // [SN]
                float*       __restrict__ out) {  // [BN, CN]
    __shared__ u16 Ah[MT * LDA];
    __shared__ u16 Al[MT * LDA];
    __shared__ u16 Bh[NT * LDA];
    __shared__ u16 Bl[NT * LDA];
    __shared__ float rinvS[NT];
    __shared__ float xinvS[MT];
    __shared__ int   clsS[NT];

    const int t    = threadIdx.x;
    const int w    = t >> 5;
    const int l    = t & 31;
    const int mBlk = blockIdx.y * MT;   // global b offset
    const int nBlk = blockIdx.x * NT;   // global s offset

    clsS[t]  = sy[nBlk + t];
    rinvS[t] = rinv[nBlk + t];
    if (t < MT) xinvS[t] = xinv[mBlk + t];

    v8f acc[8];
#pragma unroll
    for (int i = 0; i < 8; ++i)
#pragma unroll
        for (int j = 0; j < 8; ++j) acc[i][j] = 0.0f;

    const int mw  = (w & 3) * 16;   // wave M offset within block tile
    const int nw  = (w >> 2) * 128; // wave N offset within block tile
    const int kbA = (l >> 4) * 8;   // unused: A frags also use kbB
    const int kbB = (l >> 4) * 16;  // B frag K base per half-wave
    const int lc  = l & 15;

    // Prefetch registers (only the active set survives DCE).
    u16x8  pAh, pAl, pBh[4], pBl[4];
    float4 qA[2], qB[8];

    auto preload = [&](int kc) {
        if constexpr (PRE) {
            int row = t >> 2, c8 = (t & 3) * 8;  // 64 rows x 4 chunks of 8
            pAh = *(const u16x8*)(xh + (size_t)(mBlk + row) * DN + kc + c8);
            pAl = *(const u16x8*)(xl + (size_t)(mBlk + row) * DN + kc + c8);
#pragma unroll
            for (int it = 0; it < 4; ++it) {
                int g = t + 256 * it, r = g >> 2, d8 = (g & 3) * 8;
                pBh[it] = *(const u16x8*)(sxh + (size_t)(nBlk + r) * DN + kc + d8);
                pBl[it] = *(const u16x8*)(sxl + (size_t)(nBlk + r) * DN + kc + d8);
            }
        } else {
#pragma unroll
            for (int it = 0; it < 2; ++it) {
                int f = t + 256 * it, row = f >> 3, c4 = (f & 7) * 4;
                qA[it] = *(const float4*)(x + (size_t)(mBlk + row) * DN + kc + c4);
            }
#pragma unroll
            for (int it = 0; it < 8; ++it) {
                int f = t + 256 * it, row = f >> 3, c4 = (f & 7) * 4;
                qB[it] = *(const float4*)(sx + (size_t)(nBlk + row) * DN + kc + c4);
            }
        }
    };
    auto stage = [&]() {
        if constexpr (PRE) {
            int row = t >> 2, c8 = (t & 3) * 8;
            *(u16x8*)(Ah + row * LDA + c8) = pAh;
            *(u16x8*)(Al + row * LDA + c8) = pAl;
#pragma unroll
            for (int it = 0; it < 4; ++it) {
                int g = t + 256 * it, r = g >> 2, d8 = (g & 3) * 8;
                *(u16x8*)(Bh + r * LDA + d8) = pBh[it];
                *(u16x8*)(Bl + r * LDA + d8) = pBl[it];
            }
        } else {
#pragma unroll
            for (int it = 0; it < 2; ++it) {
                int f = t + 256 * it, row = f >> 3, c4 = (f & 7) * 4;
                u16x4 h, lo;
                split4(qA[it], h, lo);
                *(u16x4*)(Ah + row * LDA + c4) = h;
                *(u16x4*)(Al + row * LDA + c4) = lo;
            }
#pragma unroll
            for (int it = 0; it < 8; ++it) {
                int f = t + 256 * it, row = f >> 3, c4 = (f & 7) * 4;
                u16x4 h, lo;
                split4(qB[it], h, lo);
                *(u16x4*)(Bh + row * LDA + c4) = h;
                *(u16x4*)(Bl + row * LDA + c4) = lo;
            }
        }
    };

    preload(0);
    for (int kc = 0; kc < DN; kc += KC) {
        __syncthreads();   // LDS tiles free (previous chunk consumed)
        stage();
        __syncthreads();
        if (kc + KC < DN) preload(kc + KC);  // overlap next global loads w/ WMMA

        // A fragments (16x32 bf16): lane holds row mw+lc; K contiguous, k = kbB + s (same order as B).
        const u16* arh = Ah + (mw + lc) * LDA;
        const u16* arl = Al + (mw + lc) * LDA;
        v16bf ah = frag16(arh + kbB, arh + kbB + 8);
        v16bf al = frag16(arl + kbB, arl + kbB + 8);

#pragma unroll
        for (int nt = 0; nt < 8; ++nt) {
            const u16* brh = Bh + (nw + nt * 16 + lc) * LDA + kbB;
            const u16* brl = Bl + (nw + nt * 16 + lc) * LDA + kbB;
            v16bf bh = frag16(brh, brh + 8);
            v16bf bl = frag16(brl, brl + 8);
            // dot(A,B) ~= Ahi*Bhi + Ahi*Blo + Alo*Bhi (drop lo*lo, ~1e-5 rel)
            acc[nt] = __builtin_amdgcn_wmma_f32_16x16x32_bf16(
                false, ah, false, bh, (short)0, acc[nt], false, false);
            acc[nt] = __builtin_amdgcn_wmma_f32_16x16x32_bf16(
                false, ah, false, bl, (short)0, acc[nt], false, false);
            acc[nt] = __builtin_amdgcn_wmma_f32_16x16x32_bf16(
                false, al, false, bh, (short)0, acc[nt], false, false);
        }
    }

    // Epilogue: d = dot*rinv; score = |d|*d*xinv = |t|*t*rinv^2*xinv,
    // scattered into out[b, sy[s]] with hardware f32 atomics.
    const int lrow = (l >> 4) * 8;  // C/D layout: lanes 16-31 hold M+8
#pragma unroll
    for (int nt = 0; nt < 8; ++nt) {
        int   sLoc = nw + nt * 16 + lc;
        float rv   = rinvS[sLoc];
        float rv2  = rv * rv;
        int   cls  = clsS[sLoc];
#pragma unroll
        for (int r = 0; r < 8; ++r) {
            int   mLoc = mw + lrow + r;
            float tt   = acc[nt][r];
            float sc   = fabsf(tt) * tt * rv2 * xinvS[mLoc];
            unsafeAtomicAdd(out + (size_t)(mBlk + mLoc) * CN + cls, sc);
        }
    }
}

extern "C" void kernel_launch(void* const* d_in, const int* in_sizes, int n_in,
                              void* d_out, int out_size, void* d_ws, size_t ws_size,
                              hipStream_t stream) {
    const float* x  = (const float*)d_in[0];
    const float* sx = (const float*)d_in[1];
    const int*   sy = (const int*)d_in[2];
    float* out = (float*)d_out;

    const size_t planeS = (size_t)SN * DN;
    const size_t planeB = (size_t)BN * DN;
    const size_t need   = (2 * planeS + 2 * planeB) * sizeof(u16)
                        + (size_t)(SN + BN) * sizeof(float);

    SIC_zero_kernel<<<(BN * CN + 255) / 256, 256, 0, stream>>>(out, BN * CN);
    dim3 grid(SN / NT, BN / MT);  // 64 x 8 = 512 blocks

    if (ws_size >= need) {
        // Fast path: pre-split bf16 hi/lo planes once; GEMM stages pure copies.
        u16* sxh = (u16*)d_ws;
        u16* sxl = sxh + planeS;
        u16* xh  = sxl + planeS;
        u16* xl  = xh + planeB;
        float* rinv = (float*)(xl + planeB);
        float* xinv = rinv + SN;
        SIC_prep_kernel<<<SN + BN, 256, 0, stream>>>(x, sx, xh, xl, sxh, sxl,
                                                     rinv, xinv);
        SIC_gemm_kernel<true><<<grid, 256, 0, stream>>>(
            x, sx, xh, xl, sxh, sxl, rinv, xinv, sy, out);
    } else {
        // Fallback: norms only in ws; GEMM splits f32 -> bf16 on the fly.
        float* rinv = (float*)d_ws;
        float* xinv = rinv + SN;
        SIC_norms_kernel<<<SN + BN, 256, 0, stream>>>(x, sx, rinv, xinv);
        SIC_gemm_kernel<false><<<grid, 256, 0, stream>>>(
            x, sx, nullptr, nullptr, nullptr, nullptr, rinv, xinv, sy, out);
    }
}
